// MPNNLayer_67886253081357
// MI455X (gfx1250) — matrix operand order repaired and checked by reading, not compiled, not run on hardware
//
#include <hip/hip_runtime.h>
#include <hip/hip_bf16.h>

#define ND 128
#define ED 64
#define HD 128
#define K1 (2*ND+ED)   /* 320 = msg MLP input dim  */
#define KU (ND+HD)     /* 256 = upd MLP input dim  */

typedef __attribute__((ext_vector_type(16))) __bf16 v16bf;
typedef __attribute__((ext_vector_type(8)))  float  v8f;

// ---------------------------------------------------------------------------
// WMMA helpers (CDNA5 wave32, v_wmma_f32_16x16x32_bf16)
// ---------------------------------------------------------------------------
__device__ __forceinline__ v8f wmma_bf16(v16bf a, v16bf b, v8f c) {
  // (neg_a, A, neg_b, B, c_mod, C, reuse_a, reuse_b)
  return __builtin_amdgcn_wmma_f32_16x16x32_bf16(false, a, false, b, (short)0, c,
                                                 false, false);
}

// A-matrix 16x32 bf16 fragment from a row-major LDS tile.
// ISA layout: lane<16 holds row=lane, K = k0+[0..7] then k0+16+[0..7];
//             lane>=16 holds row=lane-16, K = k0+8+[0..7] then k0+24+[0..7].
__device__ __forceinline__ v16bf load_a_frag(const __bf16* row_ptr, int k0, int hi) {
  union { v16bf v; uint4 q[2]; } u;
  u.q[0] = *(const uint4*)(row_ptr + k0 + hi * 8);
  u.q[1] = *(const uint4*)(row_ptr + k0 + 16 + hi * 8);
  return u.v;
}

// B-matrix 32x16 bf16 fragment from column-major (N-major) bf16 weights wT[n][k].
// Lane holds N = lane&15; K = k0 + 8*(lane>=16) + [0..7] then k0 + 16 + 8*(lane>=16) + [0..7].
__device__ __forceinline__ v16bf load_b_frag(const __bf16* wT, int Kdim, int n,
                                             int k0, int hi) {
  const __bf16* p = wT + (size_t)n * Kdim + k0 + hi * 8;
  union { v16bf v; uint4 q[2]; } u;
  u.q[0] = *(const uint4*)(p);
  u.q[1] = *(const uint4*)(p + 16);
  return u.v;
}

// ---------------------------------------------------------------------------
// Weight prep: f32 (K x N, row-major) -> bf16 (N x K, row-major) transpose
// ---------------------------------------------------------------------------
__global__ void wconv_kernel(const float* __restrict__ w, __bf16* __restrict__ wT,
                             int K, int Nn) {
  int total = K * Nn;
  for (int idx = blockIdx.x * blockDim.x + threadIdx.x; idx < total;
       idx += gridDim.x * blockDim.x) {
    int k = idx / Nn, n = idx % Nn;
    wT[(size_t)n * K + k] = (__bf16)w[idx];
  }
}

// ---------------------------------------------------------------------------
// Edge message kernel: 32 edges/block, 8 waves, fused GEMM+LN+ReLU+GEMM+scatter
// ---------------------------------------------------------------------------
__global__ __launch_bounds__(256)
void edge_msg_kernel(const float* __restrict__ node_feats,
                     const float* __restrict__ edge_feats,
                     const int*   __restrict__ edge_index,
                     const __bf16* __restrict__ w1T,
                     const float* __restrict__ b1,
                     const float* __restrict__ g1,
                     const float* __restrict__ be1,
                     const __bf16* __restrict__ w2T,
                     const float* __restrict__ b2,
                     float* __restrict__ agg,
                     int E, int twoE) {
  __shared__ __bf16 A1[32][K1];   // gathered inputs, bf16   (20 KB)
  __shared__ float  Hs[32][HD];   // hidden pre-LN, f32      (16 KB)
  __shared__ __bf16 A2[32][HD];   // post-ReLU, bf16         ( 8 KB)
  __shared__ int src_s[32], tgt_s[32], eix_s[32];

  const int tid  = threadIdx.x;
  const int base = blockIdx.x * 32;

  if (tid < 32) {
    int e = base + tid;
    int s = -1, t = -1, ei = 0;
    if (e < twoE) {
      if (e < E) { s = edge_index[e];      t = edge_index[E + e]; ei = e; }
      else       { int e2 = e - E;
                   s = edge_index[E + e2]; t = edge_index[e2];    ei = e2; }
    }
    src_s[tid] = s; tgt_s[tid] = t; eix_s[tid] = ei;
  }
  __syncthreads();

  // Gather [x_src | x_tgt | e_feat] -> bf16 rows of 320 (80 float4 per row)
  for (int idx = tid; idx < 32 * 80; idx += 256) {
    int row = idx / 80, q = idx % 80;
    float4 v = make_float4(0.f, 0.f, 0.f, 0.f);
    int s = src_s[row];
    if (s >= 0) {
      if (q < 32)      v = ((const float4*)(node_feats + (size_t)s * ND))[q];
      else if (q < 64) v = ((const float4*)(node_feats + (size_t)tgt_s[row] * ND))[q - 32];
      else             v = ((const float4*)(edge_feats + (size_t)eix_s[row] * ED))[q - 64];
    }
    __bf16* p = &A1[row][q * 4];
    p[0] = (__bf16)v.x; p[1] = (__bf16)v.y; p[2] = (__bf16)v.z; p[3] = (__bf16)v.w;
  }
  __syncthreads();

  const int wv   = tid >> 5;        // wave id -> 16-column slab
  const int lane = tid & 31;
  const int hi   = lane >> 4;
  const int lr   = lane & 15;
  const int colg = wv * 16 + lr;    // output column (also C-layout column)

  // GEMM1: (32 x 320) x (320 x 128); each wave: 2 row-tiles, B reused 2x
  v8f acc0 = {0,0,0,0,0,0,0,0}, acc1 = {0,0,0,0,0,0,0,0};
  for (int k0 = 0; k0 < K1; k0 += 32) {
    v16bf bm = load_b_frag(w1T, K1, colg, k0, hi);
    v16bf a0 = load_a_frag(&A1[lr][0],      k0, hi);
    v16bf a1 = load_a_frag(&A1[16 + lr][0], k0, hi);
    acc0 = wmma_bf16(a0, bm, acc0);
    acc1 = wmma_bf16(a1, bm, acc1);
  }
  {
    float bv = b1[colg];
#pragma unroll
    for (int g = 0; g < 8; ++g) {      // C layout: row = g + 8*hi, col = lane&15
      Hs[g + hi * 8][colg]      = acc0[g] + bv;
      Hs[16 + g + hi * 8][colg] = acc1[g] + bv;
    }
  }
  __syncthreads();

  // LayerNorm (over 128) + ReLU -> bf16.  8 threads per row, shfl butterfly.
  {
    int row = tid >> 3, j = tid & 7;
    float s = 0.f, ss = 0.f;
#pragma unroll
    for (int i = 0; i < 16; ++i) { float x = Hs[row][j + 8 * i]; s += x; ss += x * x; }
#pragma unroll
    for (int m = 1; m < 8; m <<= 1) { s += __shfl_xor(s, m, 32); ss += __shfl_xor(ss, m, 32); }
    float mean = s * (1.f / HD);
    float var  = ss * (1.f / HD) - mean * mean;
    float rstd = rsqrtf(var + 1e-5f);
#pragma unroll
    for (int i = 0; i < 16; ++i) {
      int c = j + 8 * i;
      float x = (Hs[row][c] - mean) * rstd * g1[c] + be1[c];
      A2[row][c] = (__bf16)fmaxf(x, 0.f);
    }
  }
  __syncthreads();

  // GEMM2: (32 x 128) x (128 x 128)
  v8f o0 = {0,0,0,0,0,0,0,0}, o1 = {0,0,0,0,0,0,0,0};
  for (int k0 = 0; k0 < HD; k0 += 32) {
    v16bf bm = load_b_frag(w2T, HD, colg, k0, hi);
    v16bf a0 = load_a_frag(&A2[lr][0],      k0, hi);
    v16bf a1 = load_a_frag(&A2[16 + lr][0], k0, hi);
    o0 = wmma_bf16(a0, bm, o0);
    o1 = wmma_bf16(a1, bm, o1);
  }
  // Scatter-add into agg (L2-resident, f32 atomics)
  {
    float bv = b2[colg];
#pragma unroll
    for (int g = 0; g < 8; ++g) {
      int r0 = g + hi * 8, r1 = 16 + g + hi * 8;
      int t0 = tgt_s[r0],  t1 = tgt_s[r1];
      if (t0 >= 0) atomicAdd(&agg[(size_t)t0 * HD + colg], o0[g] + bv);
      if (t1 >= 0) atomicAdd(&agg[(size_t)t1 * HD + colg], o1[g] + bv);
    }
  }
}

// ---------------------------------------------------------------------------
// Node update kernel: 32 nodes/block, fused GEMM+LN+ReLU+GEMM+residual
// ---------------------------------------------------------------------------
__global__ __launch_bounds__(256)
void node_upd_kernel(const float* __restrict__ node_feats,
                     const float* __restrict__ agg,
                     const __bf16* __restrict__ w1T,
                     const float* __restrict__ b1,
                     const float* __restrict__ g1,
                     const float* __restrict__ be1,
                     const __bf16* __restrict__ w2T,
                     const float* __restrict__ b2,
                     float* __restrict__ out,
                     int N) {
  __shared__ __bf16 A1[32][KU];   // [x | agg], bf16 (16 KB)
  __shared__ float  Hs[32][HD];   // (16 KB)
  __shared__ __bf16 A2[32][HD];   // ( 8 KB)

  const int tid  = threadIdx.x;
  const int base = blockIdx.x * 32;

  for (int idx = tid; idx < 32 * 64; idx += 256) {   // 64 float4 per row
    int row = idx >> 6, q = idx & 63;
    int gr = base + row;
    float4 v = make_float4(0.f, 0.f, 0.f, 0.f);
    if (gr < N) {
      v = (q < 32) ? ((const float4*)(node_feats + (size_t)gr * ND))[q]
                   : ((const float4*)(agg        + (size_t)gr * HD))[q - 32];
    }
    __bf16* p = &A1[row][q * 4];
    p[0] = (__bf16)v.x; p[1] = (__bf16)v.y; p[2] = (__bf16)v.z; p[3] = (__bf16)v.w;
  }
  __syncthreads();

  const int wv   = tid >> 5;
  const int lane = tid & 31;
  const int hi   = lane >> 4;
  const int lr   = lane & 15;
  const int colg = wv * 16 + lr;

  v8f acc0 = {0,0,0,0,0,0,0,0}, acc1 = {0,0,0,0,0,0,0,0};
  for (int k0 = 0; k0 < KU; k0 += 32) {
    v16bf bm = load_b_frag(w1T, KU, colg, k0, hi);
    v16bf a0 = load_a_frag(&A1[lr][0],      k0, hi);
    v16bf a1 = load_a_frag(&A1[16 + lr][0], k0, hi);
    acc0 = wmma_bf16(a0, bm, acc0);
    acc1 = wmma_bf16(a1, bm, acc1);
  }
  {
    float bv = b1[colg];
#pragma unroll
    for (int g = 0; g < 8; ++g) {
      Hs[g + hi * 8][colg]      = acc0[g] + bv;
      Hs[16 + g + hi * 8][colg] = acc1[g] + bv;
    }
  }
  __syncthreads();

  {
    int row = tid >> 3, j = tid & 7;
    float s = 0.f, ss = 0.f;
#pragma unroll
    for (int i = 0; i < 16; ++i) { float x = Hs[row][j + 8 * i]; s += x; ss += x * x; }
#pragma unroll
    for (int m = 1; m < 8; m <<= 1) { s += __shfl_xor(s, m, 32); ss += __shfl_xor(ss, m, 32); }
    float mean = s * (1.f / HD);
    float var  = ss * (1.f / HD) - mean * mean;
    float rstd = rsqrtf(var + 1e-5f);
#pragma unroll
    for (int i = 0; i < 16; ++i) {
      int c = j + 8 * i;
      float x = (Hs[row][c] - mean) * rstd * g1[c] + be1[c];
      A2[row][c] = (__bf16)fmaxf(x, 0.f);
    }
  }
  __syncthreads();

  v8f o0 = {0,0,0,0,0,0,0,0}, o1 = {0,0,0,0,0,0,0,0};
  for (int k0 = 0; k0 < HD; k0 += 32) {
    v16bf bm = load_b_frag(w2T, HD, colg, k0, hi);
    v16bf a0 = load_a_frag(&A2[lr][0],      k0, hi);
    v16bf a1 = load_a_frag(&A2[16 + lr][0], k0, hi);
    o0 = wmma_bf16(a0, bm, o0);
    o1 = wmma_bf16(a1, bm, o1);
  }
  {
    float bv = b2[colg];
#pragma unroll
    for (int g = 0; g < 8; ++g) {
      int r0 = g + hi * 8, gr0 = base + r0;
      int r1 = 16 + g + hi * 8, gr1 = base + r1;
      if (gr0 < N)
        out[(size_t)gr0 * ND + colg] = node_feats[(size_t)gr0 * ND + colg] + o0[g] + bv;
      if (gr1 < N)
        out[(size_t)gr1 * ND + colg] = node_feats[(size_t)gr1 * ND + colg] + o1[g] + bv;
    }
  }
}

// ---------------------------------------------------------------------------
extern "C" void kernel_launch(void* const* d_in, const int* in_sizes, int n_in,
                              void* d_out, int out_size, void* d_ws, size_t ws_size,
                              hipStream_t stream) {
  const float* node_feats = (const float*)d_in[0];
  const float* edge_feats = (const float*)d_in[1];
  const int*   edge_index = (const int*)d_in[2];
  const float* msg_w1 = (const float*)d_in[3];
  const float* msg_b1 = (const float*)d_in[4];
  const float* ln1_g  = (const float*)d_in[5];
  const float* ln1_b  = (const float*)d_in[6];
  const float* msg_w2 = (const float*)d_in[7];
  const float* msg_b2 = (const float*)d_in[8];
  const float* upd_w1 = (const float*)d_in[9];
  const float* upd_b1 = (const float*)d_in[10];
  const float* ln2_g  = (const float*)d_in[11];
  const float* ln2_b  = (const float*)d_in[12];
  const float* upd_w2 = (const float*)d_in[13];
  const float* upd_b2 = (const float*)d_in[14];

  const int N    = in_sizes[0] / ND;
  const int E    = in_sizes[2] / 2;
  const int twoE = 2 * E;

  // Workspace layout
  char* ws = (char*)d_ws;
  size_t off = 0;
  float* agg = (float*)(ws + off);
  off += ((size_t)N * HD * sizeof(float) + 255) & ~(size_t)255;
  __bf16* w1T = (__bf16*)(ws + off); off += ((size_t)HD * K1 * 2 + 255) & ~(size_t)255;
  __bf16* w2T = (__bf16*)(ws + off); off += ((size_t)HD * HD * 2 + 255) & ~(size_t)255;
  __bf16* u1T = (__bf16*)(ws + off); off += ((size_t)HD * KU * 2 + 255) & ~(size_t)255;
  __bf16* u2T = (__bf16*)(ws + off); off += ((size_t)HD * HD * 2 + 255) & ~(size_t)255;

  hipMemsetAsync(agg, 0, (size_t)N * HD * sizeof(float), stream);

  wconv_kernel<<<80, 256, 0, stream>>>(msg_w1, w1T, K1, HD);
  wconv_kernel<<<80, 256, 0, stream>>>(msg_w2, w2T, HD, HD);
  wconv_kernel<<<80, 256, 0, stream>>>(upd_w1, u1T, KU, HD);
  wconv_kernel<<<80, 256, 0, stream>>>(upd_w2, u2T, HD, HD);

  int edgeBlocks = (twoE + 31) / 32;
  edge_msg_kernel<<<edgeBlocks, 256, 0, stream>>>(
      node_feats, edge_feats, edge_index,
      w1T, msg_b1, ln1_g, ln1_b, w2T, msg_b2, agg, E, twoE);

  int nodeBlocks = (N + 31) / 32;
  node_upd_kernel<<<nodeBlocks, 256, 0, stream>>>(
      node_feats, agg, u1T, upd_b1, ln2_g, ln2_b, u2T, upd_b2,
      (float*)d_out, N);
}
